// CausalGraphGenerator_8014408974622
// MI455X (gfx1250) — matrix operand order repaired and checked by reading, not compiled, not run on hardware
//
#include <hip/hip_runtime.h>

typedef _Float16 v16h __attribute__((ext_vector_type(16)));
typedef _Float16 v8h  __attribute__((ext_vector_type(8)));
typedef float    v8f  __attribute__((ext_vector_type(8)));
typedef float    v4f  __attribute__((ext_vector_type(4)));

#define N_NODES 8192
#define D_EMB   64
#define K_NEIGH 12

__device__ __forceinline__ float fast_tanh(float y) {
#if __has_builtin(__builtin_amdgcn_tanhf)
    return __builtin_amdgcn_tanhf(y);
#else
    float e = __expf(2.0f * y);
    return 1.0f - 2.0f * __builtin_amdgcn_rcpf(e + 1.0f);
#endif
}

// monotonic float -> uint key (bigger key == bigger float)
__device__ __forceinline__ unsigned int fkey(float t) {
    unsigned int b = __float_as_uint(t);
    return (b & 0x80000000u) ? ~b : (b | 0x80000000u);
}
__device__ __forceinline__ float unfkey(unsigned int u) {
    unsigned int b = (u & 0x80000000u) ? (u & 0x7FFFFFFFu) : ~u;
    return __uint_as_float(b);
}

// erf(z) for |z| <= 0.7072 : degree-6 Taylor in z^2, abs err ~1e-7
__device__ __forceinline__ float erf_small(float z) {
    float z2 = z * z;
    float p = 0.00012055332981789664f;           // 2/sqrt(pi)/9360
    p = fmaf(p, z2, -0.00085483270234508528f);   // /1320
    p = fmaf(p, z2,  0.0052239776254421878f);    // /216
    p = fmaf(p, z2, -0.026866170645131252f);     // /42
    p = fmaf(p, z2,  0.11283791670955126f);      // /10
    p = fmaf(p, z2, -0.37612638903183752f);      // /3
    p = fmaf(p, z2,  1.1283791670955126f);       // 2/sqrt(pi)
    return z * p;
}

__device__ __forceinline__ float gelu_exact(float x) {
    return 0.5f * x * (1.0f + erf_small(x * 0.70710678118654752f));
}

__device__ __forceinline__ void topk_insert(unsigned long long (&keys)[K_NEIGH],
                                            unsigned long long &kmin,
                                            unsigned long long key) {
    if (key <= kmin) return;
    int am = 0;
    unsigned long long mv = keys[0];
#pragma unroll
    for (int i = 1; i < K_NEIGH; ++i) if (keys[i] < mv) { mv = keys[i]; am = i; }
#pragma unroll
    for (int i = 0; i < K_NEIGH; ++i) if (i == am) keys[i] = key;
    mv = keys[0];
#pragma unroll
    for (int i = 1; i < K_NEIGH; ++i) if (keys[i] < mv) mv = keys[i];
    kmin = mv;
}

// ---------------------------------------------------------------------------
// Kernel 1: nodevec = tanh(3*(emb @ lin^T + b)), stored as f16 for WMMA.
// ---------------------------------------------------------------------------
__global__ __launch_bounds__(256) void nodevec_kernel(const float* __restrict__ emb,
                                                      const float* __restrict__ lin,
                                                      const float* __restrict__ bias,
                                                      _Float16* __restrict__ outh) {
    int n = blockIdx.x * 4 + threadIdx.y;
    int d = threadIdx.x;
    const v4f* er = (const v4f*)(emb + (size_t)n * D_EMB);
    const v4f* lr = (const v4f*)(lin + (size_t)d * D_EMB);
    float s = 0.0f;
#pragma unroll
    for (int k = 0; k < D_EMB / 4; ++k) {
        v4f a = er[k], b = lr[k];
        s = fmaf(a[0], b[0], s);
        s = fmaf(a[1], b[1], s);
        s = fmaf(a[2], b[2], s);
        s = fmaf(a[3], b[3], s);
    }
    float t = fast_tanh(3.0f * (s + bias[d]));
    outh[(size_t)n * D_EMB + d] = (_Float16)t;
}

// ---------------------------------------------------------------------------
// Kernel 2 (fused): WMMA f16 GEMM strip (16 rows x 8192 cols) -> tanh ->
// streaming per-row top-12 -> exact-gelu on winners -> masked output write.
// Block = 128 threads (4 waves). Wave w handles column tiles [w*128, w*128+128).
// B fragments are software-pipelined (double-buffered) so the L2 load latency
// of tile t+1 is hidden behind the tanh/top-k work of tile t.
// ---------------------------------------------------------------------------
__global__ __launch_bounds__(128) void adj_topk_kernel(const _Float16* __restrict__ nv1h,
                                                       const _Float16* __restrict__ nv2h,
                                                       float* __restrict__ out) {
    __shared__ float staged[4][16 * 20];                    // per-wave stage, pad 20
    __shared__ unsigned long long mergeK[4][32][K_NEIGH];   // partial top-k
    __shared__ unsigned int rowmask[16][N_NODES / 32];      // 16 KB bitmask
    __shared__ int   selCol[16][K_NEIGH];
    __shared__ float selVal[16][K_NEIGH];

    const int tid  = threadIdx.x;
    const int wave = tid >> 5;
    const int lane = tid & 31;
    const int grp  = lane >> 4;   // 0 or 1
    const int lm   = lane & 15;
    const int rowBase = blockIdx.x * 16;

    // --- A fragments: 16 rows x K=64 of nv1 (contiguous k = 16*grp + s per 32-k chunk) ---
    union { v16h v; v8h h[2]; } A0, A1;
    const _Float16* arow = nv1h + (size_t)(rowBase + lm) * D_EMB;
    const int c0 = 16 * grp;
    A0.h[0] = *(const v8h*)(arow + c0);
    A0.h[1] = *(const v8h*)(arow + c0 + 8);
    A1.h[0] = *(const v8h*)(arow + c0 + 32);
    A1.h[1] = *(const v8h*)(arow + c0 + 40);

    unsigned long long keys[K_NEIGH];
#pragma unroll
    for (int i = 0; i < K_NEIGH; ++i) keys[i] = 0ull;
    unsigned long long kmin = 0ull;

    float* stg = &staged[wave][0];
    const int t0 = wave * 128, t1 = t0 + 128;

    // per-lane base pointer into nv2 (column = tile*16 + lm, K-chunk = 16*grp)
    const _Float16* bbase = nv2h + (size_t)lm * D_EMB + 16 * grp;

    // prologue: load tile t0's B fragments
    v16h B0 = *(const v16h*)(bbase + (size_t)t0 * 16 * D_EMB);
    v16h B1 = *(const v16h*)(bbase + (size_t)t0 * 16 * D_EMB + 32);

    for (int tile = t0; tile < t1; ++tile) {
        // issue next tile's loads early (uniform wrap keeps EXEC all-ones)
        const int tn = (tile + 1 < t1) ? (tile + 1) : t0;
        v16h B0n = *(const v16h*)(bbase + (size_t)tn * 16 * D_EMB);
        v16h B1n = *(const v16h*)(bbase + (size_t)tn * 16 * D_EMB + 32);

        v8f c = {};
        c = __builtin_amdgcn_wmma_f32_16x16x32_f16(false, A0.v, false, B0, (short)0, c, false, false);
        c = __builtin_amdgcn_wmma_f32_16x16x32_f16(false, A1.v, false, B1, (short)0, c, false, false);

        // tanh(3a) and stage: value (r, lm-col) belongs to row r + 8*grp
#pragma unroll
        for (int r = 0; r < 8; ++r) {
            float t = fast_tanh(3.0f * c[r]);
            stg[(r + 8 * grp) * 20 + lm] = t;
        }
        asm volatile("s_wait_dscnt 0" ::: "memory");

        // handler: lane processes row lm, cols [8*grp, 8*grp+8) of this tile
        v4f q0 = *(v4f*)(stg + lm * 20 + 8 * grp);
        v4f q1 = *(v4f*)(stg + lm * 20 + 8 * grp + 4);
        const int colb = tile * 16 + 8 * grp;
#pragma unroll
        for (int j = 0; j < 8; ++j) {
            float tv = (j < 4) ? q0[j] : q1[j - 4];
            unsigned long long key =
                ((unsigned long long)fkey(tv) << 32) | (unsigned int)(8191 - (colb + j));
            topk_insert(keys, kmin, key);
        }

        B0 = B0n;
        B1 = B1n;
    }

    // publish partial top-k, clear bitmask
#pragma unroll
    for (int i = 0; i < K_NEIGH; ++i) mergeK[wave][lane][i] = keys[i];
    for (int i = tid; i < 16 * (N_NODES / 32); i += 128)
        ((unsigned int*)rowmask)[i] = 0u;
    __syncthreads();

    // merge: thread r owns row r; 4 waves x 2 half-groups x 12 = 96 candidates
    if (tid < 16) {
        unsigned long long best[K_NEIGH];
#pragma unroll
        for (int i = 0; i < K_NEIGH; ++i) best[i] = 0ull;
        unsigned long long bmin = 0ull;
        for (int w = 0; w < 4; ++w)
            for (int g = 0; g < 2; ++g) {
                int src = g * 16 + tid;
                for (int i = 0; i < K_NEIGH; ++i)
                    topk_insert(best, bmin, mergeK[w][src][i]);
            }
        for (int i = 0; i < K_NEIGH; ++i) {
            unsigned long long key = best[i];
            int col = 8191 - (int)(key & 0x1FFFull);
            float tv = unfkey((unsigned int)(key >> 32));
            selCol[tid][i] = col;
            selVal[tid][i] = gelu_exact(tv);
            rowmask[tid][col >> 5] |= (1u << (col & 31));   // row-private, no race
        }
    }
    __syncthreads();

    // output: zeros everywhere except the 12 selected entries per row
    float* orow0 = out + (size_t)rowBase * N_NODES;
    for (int idx = tid; idx < 16 * (N_NODES / 4); idx += 128) {
        int row = idx >> 11;           // N_NODES/4 = 2048
        int col = (idx & 2047) << 2;
        unsigned int m = rowmask[row][col >> 5];
        v4f o = {0.0f, 0.0f, 0.0f, 0.0f};
        if ((m >> (col & 31)) & 0xFu) {
#pragma unroll
            for (int j = 0; j < 4; ++j) {
                if ((m >> ((col + j) & 31)) & 1u) {
                    float v = 0.0f;
#pragma unroll
                    for (int i = 0; i < K_NEIGH; ++i)
                        if (selCol[row][i] == col + j) v = selVal[row][i];
                    o[j] = v;
                }
            }
        }
        __builtin_nontemporal_store(o, (v4f*)(orow0 + (size_t)row * N_NODES + col));
    }
}

extern "C" void kernel_launch(void* const* d_in, const int* in_sizes, int n_in,
                              void* d_out, int out_size, void* d_ws, size_t ws_size,
                              hipStream_t stream) {
    const float* emb1  = (const float*)d_in[0];
    const float* emb2  = (const float*)d_in[1];
    const float* lin1w = (const float*)d_in[2];
    const float* lin1b = (const float*)d_in[3];
    const float* lin2w = (const float*)d_in[4];
    const float* lin2b = (const float*)d_in[5];
    float* out = (float*)d_out;

    _Float16* nv1h = (_Float16*)d_ws;
    _Float16* nv2h = (_Float16*)((char*)d_ws + (size_t)N_NODES * D_EMB * sizeof(_Float16));

    dim3 b1(64, 4);
    nodevec_kernel<<<N_NODES / 4, b1, 0, stream>>>(emb1, lin1w, lin1b, nv1h);
    nodevec_kernel<<<N_NODES / 4, b1, 0, stream>>>(emb2, lin2w, lin2b, nv2h);

    adj_topk_kernel<<<N_NODES / 16, 128, 0, stream>>>(nv1h, nv2h, out);
}
